// ResidualVectorQuantization_5068061409927
// MI455X (gfx1250) — matrix operand order repaired and checked by reading, not compiled, not run on hardware
//
#include <hip/hip_runtime.h>
#include <hip/hip_bf16.h>

typedef __attribute__((ext_vector_type(16))) __bf16 v16bf;
typedef __attribute__((ext_vector_type(8)))  float  v8f;

#define B_DIM   16
#define N_DIM   4096
#define D_DIM   512
#define K_CB    1024
#define NUM_Q   8
#define ROWS    (B_DIM * N_DIM)          // 65536 residual rows

#define BS_PAD    8
#define BS_STRIDE (D_DIM + BS_PAD)       // 520 bf16 -> 1040B row stride (16B aligned, bank-rotating)
#define CHUNK     16                     // codes per LDS tile (double buffered)
#define NCHUNK    (K_CB / CHUNK)         // 64

// workspace layout (bytes)
#define RESID_OFF 0ull                                        // fp32 residual [ROWS][D]   = 134217728 B
#define CB_OFF    ((size_t)ROWS * D_DIM * 4)                  // bf16 codebooks [Q][K][D]  = 8388608 B
#define EN_OFF    (CB_OFF + (size_t)NUM_Q * K_CB * D_DIM * 2) // fp32 ||e||^2 [Q][K]       = 32768 B

// ---------------------------------------------------------------------------
// CDNA5 async global->LDS copy (ASYNCcnt-tracked DMA, no VGPR round trip).
// vdst = 32-bit LDS byte address (low 32 bits of generic LDS pointer),
// vaddr = 64-bit global address, saddr = off.
// ---------------------------------------------------------------------------
__device__ __forceinline__ void async_ld_b128(unsigned int lds_addr, const void* gaddr)
{
    asm volatile("global_load_async_to_lds_b128 %0, %1, off"
                 :: "v"(lds_addr), "v"(gaddr) : "memory");
}
__device__ __forceinline__ void wait_async0()
{
    asm volatile("s_wait_asynccnt 0" ::: "memory");
}

// ---------------------------------------------------------------------------
// Codebook prep: fp32 -> bf16 copy + squared-norm per code row.
// ---------------------------------------------------------------------------
__global__ void cb_prep(const float* __restrict__ cb,
                        __bf16* __restrict__ cbb,
                        float* __restrict__ enorm)
{
    const long row = blockIdx.x;                 // q*1024 + k
    const float* s = cb  + row * D_DIM;
    __bf16*      d = cbb + row * D_DIM;
    float acc = 0.0f;
    for (int i = threadIdx.x; i < D_DIM; i += 128) {
        float v = s[i];
        d[i] = (__bf16)v;
        acc += v * v;
    }
    __shared__ float red[128];
    red[threadIdx.x] = acc;
    __syncthreads();
    for (int o = 64; o > 0; o >>= 1) {
        if (threadIdx.x < o) red[threadIdx.x] += red[threadIdx.x + o];
        __syncthreads();
    }
    if (threadIdx.x == 0) enorm[row] = red[0];
}

// ---------------------------------------------------------------------------
// residual init: r[b][n][d] = x[b][d][n]   (LDS tiled transpose)
// ---------------------------------------------------------------------------
__global__ void transpose_init(const float* __restrict__ x,
                               float* __restrict__ resid)
{
    __shared__ float tile[32][33];
    const int b  = blockIdx.z;
    const int d0 = blockIdx.y * 32;
    const int n0 = blockIdx.x * 32;
    const int tx = threadIdx.x & 31;
    const int ty = threadIdx.x >> 5;
#pragma unroll
    for (int i = 0; i < 32; i += 8)
        tile[ty + i][tx] = x[((long)b * D_DIM + d0 + ty + i) * N_DIM + n0 + tx];
    __syncthreads();
#pragma unroll
    for (int i = 0; i < 32; i += 8)
        resid[((long)b * N_DIM + n0 + ty + i) * D_DIM + d0 + tx] = tile[tx][ty + i];
}

// ---------------------------------------------------------------------------
// quantized_out[b][d][n] = x[b][d][n] - resid_final[b][n][d]
// ---------------------------------------------------------------------------
__global__ void final_out(const float* __restrict__ x,
                          const float* __restrict__ resid,
                          float* __restrict__ out)
{
    __shared__ float tile[32][33];
    const int b  = blockIdx.z;
    const int d0 = blockIdx.y * 32;
    const int n0 = blockIdx.x * 32;
    const int tx = threadIdx.x & 31;
    const int ty = threadIdx.x >> 5;
#pragma unroll
    for (int i = 0; i < 32; i += 8)
        tile[ty + i][tx] = resid[((long)b * N_DIM + n0 + ty + i) * D_DIM + d0 + tx];
    __syncthreads();
#pragma unroll
    for (int i = 0; i < 32; i += 8) {
        long xi = ((long)b * D_DIM + d0 + ty + i) * N_DIM + n0 + tx;
        out[xi] = x[xi] - tile[tx][ty + i];
    }
}

// ---------------------------------------------------------------------------
// Cooperative async fill of one 16-code LDS tile (16KB): 4 x b128 per thread,
// flat-coalesced across the 256 threads.
// ---------------------------------------------------------------------------
__device__ __forceinline__ void issue_chunk(const __bf16* __restrict__ cbq, int cc,
                                            __bf16* buf, int tid)
{
#pragma unroll
    for (int jj = 0; jj < 4; ++jj) {
        const int s = jj * 256 + tid;       // int4 slot 0..1023
        const int c = s >> 6;               // code row 0..15 (64 slots per row)
        const int d = (s & 63) * 8;         // bf16 element offset
        const unsigned int la = (unsigned int)(uintptr_t)&buf[c * BS_STRIDE + d];
        async_ld_b128(la, (const void*)(cbq + (long)(cc + c) * D_DIM + d));
    }
}

// ---------------------------------------------------------------------------
// One RVQ stage: fused (dist GEMM + argmin + residual update).
// Block = 256 threads (8 waves); each block owns 128 residual rows.
// Wave w holds rows [wgRow + 16w, +16) as resident bf16 A fragments (D=512).
// Codebook streamed via async-DMA double-buffered 16-code LDS tiles.
// ---------------------------------------------------------------------------
__global__ __launch_bounds__(256)
void rvq_stage(const float* __restrict__ cbf32,   // original fp32 codebooks [Q][K][D]
               const __bf16* __restrict__ cbb,    // bf16 codebooks
               const float* __restrict__ enorm,   // ||e||^2 [Q][K]
               float* __restrict__ resid,         // fp32 residual [ROWS][D] (in place)
               float* __restrict__ idx_out,       // indices as float, [Q][ROWS]
               int q)
{
    __shared__ __bf16 Bs[2][CHUNK * BS_STRIDE];   // 2 x 16.6KB double buffer
    __shared__ int rowIdx[128];

    const int tid  = threadIdx.x;
    const int wave = tid >> 5;
    const int lane = tid & 31;
    const int lh   = lane & 15;
    const int hi   = lane >> 4;                   // 16-lane half
    const long wgRow = (long)blockIdx.x * 128;

    // --- A fragments: 16 rows x 512 deep, fp32 -> bf16, contiguous k = 16*hi + e layout ---
    const float* rp = resid + (wgRow + wave * 16 + lh) * (long)D_DIM;
    v16bf afrag[16];
#pragma unroll
    for (int kd = 0; kd < 16; ++kd) {
        const float* p = rp + kd * 32 + hi * 16;
        v16bf a;
#pragma unroll
        for (int e = 0; e < 8; ++e) a[e] = (__bf16)p[e];
#pragma unroll
        for (int e = 0; e < 8; ++e) a[e + 8] = (__bf16)p[8 + e];
        afrag[kd] = a;
    }

    float minv[8];
    int   mini[8];
#pragma unroll
    for (int j = 0; j < 8; ++j) { minv[j] = 3.4e38f; mini[j] = 0; }

    const __bf16* cbq = cbb + (long)q * K_CB * D_DIM;
    const float*  enq = enorm + q * K_CB;

    // prologue: fill buf0, start buf1
    issue_chunk(cbq, 0, Bs[0], tid);
    wait_async0();
    __syncthreads();
    issue_chunk(cbq, CHUNK, Bs[1], tid);

    for (int i = 0; i < NCHUNK; ++i) {
        const __bf16* brow = &Bs[i & 1][lh * BS_STRIDE + hi * 16];
        // two independent accumulation chains (even/odd K-steps) for WMMA ILP
        v8f c0 = {}, c1 = {};
#pragma unroll
        for (int kd = 0; kd < 16; kd += 2) {
            v16bf b0 = *(const v16bf*)(brow + kd * 32);
            v16bf b1 = *(const v16bf*)(brow + (kd + 1) * 32);
            c0 = __builtin_amdgcn_wmma_f32_16x16x32_bf16(
                     false, afrag[kd],     false, b0, (short)0, c0, false, false);
            c1 = __builtin_amdgcn_wmma_f32_16x16x32_bf16(
                     false, afrag[kd + 1], false, b1, (short)0, c1, false, false);
        }
        const int   code = i * CHUNK + lh;
        const float en   = enq[code];
#pragma unroll
        for (int j = 0; j < 8; ++j) {              // rows: M = hi*8 + j
            float sc = en - 2.0f * (c0[j] + c1[j]);
            if (sc < minv[j]) { minv[j] = sc; mini[j] = code; }
        }
        // retire chunk i+1's DMA, release buf[i&1], start chunk i+2 into it
        wait_async0();
        __syncthreads();
        if (i + 2 < NCHUNK)
            issue_chunk(cbq, (i + 2) * CHUNK, Bs[i & 1], tid);
    }

    // --- argmin reduce across the 16 lanes sharing each row set ---
#pragma unroll
    for (int off = 1; off < 16; off <<= 1) {
#pragma unroll
        for (int j = 0; j < 8; ++j) {
            float ov = __shfl_xor(minv[j], off, 32);
            int   oi = __shfl_xor(mini[j], off, 32);
            if (ov < minv[j] || (ov == minv[j] && oi < mini[j])) {
                minv[j] = ov; mini[j] = oi;
            }
        }
    }
    if (lh == 0) {
#pragma unroll
        for (int j = 0; j < 8; ++j)
            rowIdx[wave * 16 + hi * 8 + j] = mini[j];
    }
    __syncthreads();

    if (tid < 128)
        idx_out[(long)q * ROWS + wgRow + tid] = (float)rowIdx[tid];

    // --- residual update in fp32 from the original codebook: r -= e[best] ---
    const int r  = tid >> 1;                       // 2 threads per row
    const int c0i = (tid & 1) * 256;
    const float* ep = cbf32 + ((long)q * K_CB + rowIdx[r]) * D_DIM + c0i;
    float* rr = resid + (wgRow + r) * (long)D_DIM + c0i;
#pragma unroll 4
    for (int ci = 0; ci < 256; ci += 4) {
        float4 ev = *(const float4*)(ep + ci);
        float4 rv = *(float4*)(rr + ci);
        rv.x -= ev.x; rv.y -= ev.y; rv.z -= ev.z; rv.w -= ev.w;
        *(float4*)(rr + ci) = rv;
    }
}

// ---------------------------------------------------------------------------
extern "C" void kernel_launch(void* const* d_in, const int* in_sizes, int n_in,
                              void* d_out, int out_size, void* d_ws, size_t ws_size,
                              hipStream_t stream)
{
    const float* x  = (const float*)d_in[0];   // [B][D][N]
    const float* cb = (const float*)d_in[1];   // [Q][K][D]

    float* out     = (float*)d_out;                               // [B][D][N]
    float* idx_out = out + (size_t)B_DIM * D_DIM * N_DIM;         // [Q][B][N] as float

    char*   ws    = (char*)d_ws;
    float*  resid = (float*)(ws + RESID_OFF);
    __bf16* cbb   = (__bf16*)(ws + CB_OFF);
    float*  enorm = (float*)(ws + EN_OFF);

    // 1) codebook bf16 + norms
    cb_prep<<<NUM_Q * K_CB, 128, 0, stream>>>(cb, cbb, enorm);

    // 2) residual = x^T
    dim3 tgrid(N_DIM / 32, D_DIM / 32, B_DIM);
    transpose_init<<<tgrid, 256, 0, stream>>>(x, resid);

    // 3) 8 sequential RVQ stages (fused GEMM + argmin + update)
    for (int q = 0; q < NUM_Q; ++q)
        rvq_stage<<<ROWS / 128, 256, 0, stream>>>(cb, cbb, enorm, resid, idx_out, q);

    // 4) quantized_out = x - residual_final (transposed back)
    final_out<<<tgrid, 256, 0, stream>>>(x, resid, out);
}
